// StackedGCN_44547400794889
// MI455X (gfx1250) — compile-verified, matrix operand order checked by reading, not run on hardware
//
#include <hip/hip_runtime.h>
#include <hip/hip_bf16.h>
#include <math.h>

typedef __attribute__((ext_vector_type(16))) __bf16 v16bf;
typedef __attribute__((ext_vector_type(8)))  float  v8f;

// ---------------------------------------------------------------------------
// Degree / normalization:  deg = 1 (self loop) + sum over edges into col;
// dinv = rsqrt(deg);  norm(e) = dinv[row]*dinv[col] computed on the fly.
// ---------------------------------------------------------------------------
__global__ void k_init_deg(float* __restrict__ deg, int n) {
    int i = blockIdx.x * blockDim.x + threadIdx.x;
    if (i < n) deg[i] = 1.0f;   // self-loop
}

__global__ void k_deg_accum(const int* __restrict__ col, float* __restrict__ deg, int e) {
    int i = blockIdx.x * blockDim.x + threadIdx.x;
    if (i < e) unsafeAtomicAdd(&deg[col[i]], 1.0f);
}

__global__ void k_finalize_dinv(float* __restrict__ dinv, int n) {
    int i = blockIdx.x * blockDim.x + threadIdx.x;
    if (i < n) {
        float d = dinv[i];
        dinv[i] = (d > 0.0f) ? rsqrtf(d) : 0.0f;
    }
}

// ---------------------------------------------------------------------------
// Pack fp32 row-major W[K,Nout] into bf16 WMMA B-fragments:
// fragment (kt,nt): lane l holds column n = nt*16 + (l&15), 16 contiguous bf16
// covering k = kt*32 + {base..base+7, base+16..base+23}, base = (l<16)?0:8.
// ---------------------------------------------------------------------------
__global__ void k_pack_w(const float* __restrict__ W, __bf16* __restrict__ Wb,
                         int K, int Nout) {
    int idx = blockIdx.x * blockDim.x + threadIdx.x;
    int total = K * Nout;
    if (idx >= total) return;
    int j    = idx & 15;
    int lane = (idx >> 4) & 31;
    int t    = idx >> 9;                 // tile = kt*numNT + nt
    int numNT = Nout >> 4;
    int nt = t % numNT;
    int kt = t / numNT;
    int n  = nt * 16 + (lane & 15);
    int k  = kt * 32 + ((j < 8) ? j : j + 8) + ((lane < 16) ? 0 : 8);
    Wb[idx] = (__bf16)W[k * Nout + n];
}

// ---------------------------------------------------------------------------
// H = act(X) @ W  via v_wmma_f32_16x16x32_bf16.
// One wave32 computes a 16 x (NT*16) strip: A fragment built once per K-step
// (optionally ReLU-fused) and reused across NT independent WMMAs.
// Requires (Nout/16) % NT == 0.
// ---------------------------------------------------------------------------
template <int NT, bool RELU>
__global__ void k_gemm_wmma(const float* __restrict__ X, const __bf16* __restrict__ Wb,
                            float* __restrict__ H, int Nrows, int K, int Nout) {
    const int wave = (blockIdx.x * blockDim.x + threadIdx.x) >> 5;
    const int lane = threadIdx.x & 31;
    const int numNT    = Nout >> 4;
    const int numGroups = numNT / NT;
    const int rowTiles = (Nrows + 15) >> 4;
    if (wave >= rowTiles * numGroups) return;
    const int mt  = wave / numGroups;
    const int nt0 = (wave % numGroups) * NT;
    const int halfSel = lane >> 4;            // 0 | 1
    const int lcol    = lane & 15;

    int rA = mt * 16 + lcol;
    if (rA >= Nrows) rA = Nrows - 1;          // clamp (stores guarded per-tile)
    const float* xrow = X + (size_t)rA * K;

    v8f acc[NT];
#pragma unroll
    for (int t = 0; t < NT; ++t) acc[t] = (v8f){};

    const int kTiles = K >> 5;
    for (int kt = 0; kt < kTiles; ++kt) {
        const int kbase = kt * 32 + halfSel * 8;
        // A fragment: k = kbase + {0..7} and kbase + 16 + {0..7}
        float4 lo0 = *(const float4*)(xrow + kbase);
        float4 lo1 = *(const float4*)(xrow + kbase + 4);
        float4 hi0 = *(const float4*)(xrow + kbase + 16);
        float4 hi1 = *(const float4*)(xrow + kbase + 20);
        if (RELU) {
            lo0.x = fmaxf(lo0.x, 0.f); lo0.y = fmaxf(lo0.y, 0.f);
            lo0.z = fmaxf(lo0.z, 0.f); lo0.w = fmaxf(lo0.w, 0.f);
            lo1.x = fmaxf(lo1.x, 0.f); lo1.y = fmaxf(lo1.y, 0.f);
            lo1.z = fmaxf(lo1.z, 0.f); lo1.w = fmaxf(lo1.w, 0.f);
            hi0.x = fmaxf(hi0.x, 0.f); hi0.y = fmaxf(hi0.y, 0.f);
            hi0.z = fmaxf(hi0.z, 0.f); hi0.w = fmaxf(hi0.w, 0.f);
            hi1.x = fmaxf(hi1.x, 0.f); hi1.y = fmaxf(hi1.y, 0.f);
            hi1.z = fmaxf(hi1.z, 0.f); hi1.w = fmaxf(hi1.w, 0.f);
        }
        v16bf a;
        a[0]  = (__bf16)lo0.x; a[1]  = (__bf16)lo0.y; a[2]  = (__bf16)lo0.z; a[3]  = (__bf16)lo0.w;
        a[4]  = (__bf16)lo1.x; a[5]  = (__bf16)lo1.y; a[6]  = (__bf16)lo1.z; a[7]  = (__bf16)lo1.w;
        a[8]  = (__bf16)hi0.x; a[9]  = (__bf16)hi0.y; a[10] = (__bf16)hi0.z; a[11] = (__bf16)hi0.w;
        a[12] = (__bf16)hi1.x; a[13] = (__bf16)hi1.y; a[14] = (__bf16)hi1.z; a[15] = (__bf16)hi1.w;

        const __bf16* wb = Wb + ((((size_t)kt * numNT + nt0) * 32 + lane) << 4);
#pragma unroll
        for (int t = 0; t < NT; ++t) {
            v16bf b = *(const v16bf*)(wb + ((size_t)t << 9));   // next tile: +32 lanes*16
            acc[t] = __builtin_amdgcn_wmma_f32_16x16x32_bf16(
                         false, a, false, b, (short)0, acc[t], false, false);
        }
    }

    // Store: lane writes column nt0*16 + t*16 + lcol of rows mt*16+halfSel*8+r.
    const int row0 = mt * 16 + halfSel * 8;
    float* pbase = H + (size_t)row0 * Nout + nt0 * 16 + lcol;
    if (mt * 16 + 16 <= Nrows) {              // full tile: unguarded stores
        size_t roff = 0;
#pragma unroll
        for (int r = 0; r < 8; ++r) {
#pragma unroll
            for (int t = 0; t < NT; ++t) pbase[roff + t * 16] = acc[t][r];
            roff += (size_t)Nout;
        }
    } else {
        size_t roff = 0;
#pragma unroll
        for (int r = 0; r < 8; ++r) {
            if (row0 + r < Nrows) {
#pragma unroll
                for (int t = 0; t < NT; ++t) pbase[roff + t * 16] = acc[t][r];
            }
            roff += (size_t)Nout;
        }
    }
}

// ---------------------------------------------------------------------------
// Acc[i,d] = b[d] + H[i,d] * dinv[i]^2      (bias + self-loop message)
// ---------------------------------------------------------------------------
__global__ void k_init_acc(const float* __restrict__ H, const float* __restrict__ dinv,
                           const float* __restrict__ bias, float* __restrict__ Acc,
                           int Nrows, int D) {
    long long idx = (long long)blockIdx.x * blockDim.x + threadIdx.x;
    long long total = (long long)Nrows * D;
    if (idx >= total) return;
    int i = (int)(idx / D), d = (int)(idx % D);
    float di = dinv[i];
    Acc[idx] = bias[d] + H[idx] * di * di;
}

// ---------------------------------------------------------------------------
// Acc[col[e], :] += H[row[e], :] * dinv[row]*dinv[col]   (float4 per thread,
// hardware global_atomic_add_f32 via unsafeAtomicAdd)
// ---------------------------------------------------------------------------
__global__ void k_edge_scatter(const int* __restrict__ src, const int* __restrict__ dst,
                               const float* __restrict__ H, const float* __restrict__ dinv,
                               float* __restrict__ Acc, int E, int D) {
    int chunks = D >> 2;                            // float4 chunks per row
    int idx = blockIdx.x * blockDim.x + threadIdx.x;
    if (idx >= E * chunks) return;
    int e = idx / chunks;
    int q = idx - e * chunks;
    int r = src[e], c = dst[e];
    float coeff = dinv[r] * dinv[c];
    const float4 v = *(const float4*)(H + (size_t)r * D + q * 4);
    float* out = Acc + (size_t)c * D + q * 4;
    unsafeAtomicAdd(out + 0, v.x * coeff);
    unsafeAtomicAdd(out + 1, v.y * coeff);
    unsafeAtomicAdd(out + 2, v.z * coeff);
    unsafeAtomicAdd(out + 3, v.w * coeff);
}

// ---------------------------------------------------------------------------
// In-place log_softmax over D columns, one wave32 per row (D=64 -> 2/lane).
// ---------------------------------------------------------------------------
__global__ void k_log_softmax(float* __restrict__ Out, int Nrows, int D) {
    int wave = (blockIdx.x * blockDim.x + threadIdx.x) >> 5;
    int lane = threadIdx.x & 31;
    if (wave >= Nrows) return;
    float* row = Out + (size_t)wave * D;
    float m = -INFINITY;
    for (int d = lane; d < D; d += 32) m = fmaxf(m, row[d]);
    for (int o = 16; o > 0; o >>= 1)   m = fmaxf(m, __shfl_xor(m, o, 32));
    float s = 0.0f;
    for (int d = lane; d < D; d += 32) s += __expf(row[d] - m);
    for (int o = 16; o > 0; o >>= 1)   s += __shfl_xor(s, o, 32);
    float lse = __logf(s) + m;
    for (int d = lane; d < D; d += 32) row[d] -= lse;
}

// ---------------------------------------------------------------------------
static inline size_t align256(size_t x) { return (x + 255) & ~(size_t)255; }

extern "C" void kernel_launch(void* const* d_in, const int* in_sizes, int n_in,
                              void* d_out, int out_size, void* d_ws, size_t ws_size,
                              hipStream_t stream) {
    const int*   edges    = (const int*)  d_in[0];   // (2, E)
    const float* features = (const float*)d_in[1];   // (N, 128)
    const float* W0 = (const float*)d_in[2]; const float* b0 = (const float*)d_in[3];
    const float* W1 = (const float*)d_in[4]; const float* b1 = (const float*)d_in[5];
    const float* W2 = (const float*)d_in[6]; const float* b2 = (const float*)d_in[7];

    const int D_IN  = 128;
    const int D_H   = 128;
    const int D_OUT = in_sizes[7];                   // 64
    const int E = in_sizes[0] / 2;
    const int N = in_sizes[1] / D_IN;

    const int* rowIdx = edges;                       // sources
    const int* colIdx = edges + E;                   // targets

    // ---- workspace carve-up ----
    char* ws = (char*)d_ws;
    size_t off = 0;
    float* dinv = (float*)(ws + off); off += align256((size_t)N * sizeof(float));
    float* bufA = (float*)(ws + off); off += align256((size_t)N * D_H * sizeof(float)); // h_lin
    float* bufB = (float*)(ws + off); off += align256((size_t)N * D_H * sizeof(float)); // acc/x
    __bf16* Wb0 = (__bf16*)(ws + off); off += align256((size_t)D_IN * D_H  * sizeof(__bf16));
    __bf16* Wb1 = (__bf16*)(ws + off); off += align256((size_t)D_H  * D_H  * sizeof(__bf16));
    __bf16* Wb2 = (__bf16*)(ws + off); off += align256((size_t)D_H  * D_OUT* sizeof(__bf16));
    (void)ws_size; (void)n_in; (void)out_size;

    const int TB = 256;
    auto blocks = [](long long work, int tb) { return (int)((work + tb - 1) / tb); };

    // ---- degree / dinv ----
    k_init_deg     <<<blocks(N, TB), TB, 0, stream>>>(dinv, N);
    k_deg_accum    <<<blocks(E, TB), TB, 0, stream>>>(colIdx, dinv, E);
    k_finalize_dinv<<<blocks(N, TB), TB, 0, stream>>>(dinv, N);

    // ---- pack weights into WMMA B-fragment layout (bf16) ----
    k_pack_w<<<blocks((long long)D_IN * D_H,  TB), TB, 0, stream>>>(W0, Wb0, D_IN, D_H);
    k_pack_w<<<blocks((long long)D_H  * D_H,  TB), TB, 0, stream>>>(W1, Wb1, D_H,  D_H);
    k_pack_w<<<blocks((long long)D_H  * D_OUT,TB), TB, 0, stream>>>(W2, Wb2, D_H,  D_OUT);

    const int rowTiles = (N + 15) / 16;
    constexpr int NT = 4;                            // 16x64 strip per wave
    auto gemm_blocks = [&](int nout) {
        long long waves = (long long)rowTiles * ((nout / 16) / NT);
        return (int)((waves * 32 + TB - 1) / TB);
    };

    // ---- layer 0: x = features ----
    k_gemm_wmma<NT,false><<<gemm_blocks(D_H), TB, 0, stream>>>(features, Wb0, bufA, N, D_IN, D_H);
    k_init_acc    <<<blocks((long long)N * D_H, TB), TB, 0, stream>>>(bufA, dinv, b0, bufB, N, D_H);
    k_edge_scatter<<<blocks((long long)E * (D_H >> 2), TB), TB, 0, stream>>>(
        rowIdx, colIdx, bufA, dinv, bufB, E, D_H);

    // ---- layer 1: x = relu(bufB) ----
    k_gemm_wmma<NT,true><<<gemm_blocks(D_H), TB, 0, stream>>>(bufB, Wb1, bufA, N, D_H, D_H);
    k_init_acc    <<<blocks((long long)N * D_H, TB), TB, 0, stream>>>(bufA, dinv, b1, bufB, N, D_H);
    k_edge_scatter<<<blocks((long long)E * (D_H >> 2), TB), TB, 0, stream>>>(
        rowIdx, colIdx, bufA, dinv, bufB, E, D_H);

    // ---- layer 2: x = relu(bufB), accumulate straight into d_out ----
    float* out = (float*)d_out;
    k_gemm_wmma<NT,true><<<gemm_blocks(D_OUT), TB, 0, stream>>>(bufB, Wb2, bufA, N, D_H, D_OUT);
    k_init_acc    <<<blocks((long long)N * D_OUT, TB), TB, 0, stream>>>(bufA, dinv, b2, out, N, D_OUT);
    k_edge_scatter<<<blocks((long long)E * (D_OUT >> 2), TB), TB, 0, stream>>>(
        rowIdx, colIdx, bufA, dinv, out, E, D_OUT);

    // ---- log_softmax in place on d_out ----
    k_log_softmax<<<blocks((long long)N * 32, TB), TB, 0, stream>>>(out, N, D_OUT);
}
